// GroupedQueryAttention_3066606649973
// MI455X (gfx1250) — compile-verified, matrix operand order checked by reading, not run on hardware
//
#include <hip/hip_runtime.h>

#define B_   2
#define T_   2048
#define DIM_ 2048
#define NH_  32
#define NKV_ 8
#define HD_  64
#define NREP_ (NH_/NKV_)

typedef __attribute__((ext_vector_type(16))) __bf16 v16bf;
typedef __attribute__((ext_vector_type(8)))  float  v8f;
typedef __attribute__((ext_vector_type(4)))  int    v4i_;

union Frag {
    v16bf v;
    uint4 q[2];
    unsigned short s[16];
};

#ifndef __has_builtin
#define __has_builtin(x) 0
#endif
#if __has_builtin(__builtin_amdgcn_global_load_async_to_lds_b128)
#define HAVE_ASYNC_COPY 1
#else
#define HAVE_ASYNC_COPY 0
#endif
#if __has_builtin(__builtin_amdgcn_s_wait_asynccnt)
#define HAVE_WAIT_ASYNC 1
#else
#define HAVE_WAIT_ASYNC 0
#endif

__device__ __forceinline__ unsigned short f2bf(float f) {
    unsigned int u = __float_as_uint(f);
    u += 0x7FFFu + ((u >> 16) & 1u);   // round-to-nearest-even
    return (unsigned short)(u >> 16);
}
__device__ __forceinline__ float bf2f(unsigned short h) {
    return __uint_as_float(((unsigned int)h) << 16);
}

// 16-byte global -> LDS copy; async (ASYNCcnt) on CDNA5 toolchains that expose it
__device__ __forceinline__ void copy16_g2l(const unsigned short* g, unsigned short* l) {
#if HAVE_ASYNC_COPY
    __builtin_amdgcn_global_load_async_to_lds_b128(
        (__attribute__((address_space(1))) v4i_*)g,
        (__attribute__((address_space(3))) v4i_*)l, 0, 0);
#else
    *(uint4*)l = *(const uint4*)g;
#endif
}
__device__ __forceinline__ void wait_async_copies() {
#if HAVE_ASYNC_COPY
#if HAVE_WAIT_ASYNC
    __builtin_amdgcn_s_wait_asynccnt(0);
#else
    asm volatile("s_wait_asynccnt 0" ::: "memory");
#endif
#endif
}

// ---------------------------------------------------------------- elementwise f32 -> bf16
__global__ void cvt_f32_bf16(const float* __restrict__ src,
                             unsigned short* __restrict__ dst, int n) {
    int i = blockIdx.x * blockDim.x + threadIdx.x;
    if (i < n) dst[i] = f2bf(src[i]);
}

// ------------------------------------------- transpose-convert: src[K,N] f32 -> dst[N,K] bf16
__global__ void tcvt_f32_bf16(const float* __restrict__ src,
                              unsigned short* __restrict__ dst, int K, int N) {
    int i = blockIdx.x * blockDim.x + threadIdx.x;
    if (i < K * N) {
        int n = i / K, k = i - n * K;
        dst[i] = f2bf(src[(size_t)k * N + n]);
    }
}

// ------------------------------------------------------------------------------------------
// Block-cooperative bf16 GEMM:  C[M,N] = A[M,K] * Wt[N,K]^T
// Block = 256 threads = 8 waves; block tile 64(M) x 256(N); K-step 32.
// A(64x32) + B(256x32) bf16 staged in LDS via async global->LDS copies each K step;
// each wave computes a 32x64 sub-tile = 8 accumulators = 8 WMMAs per K step from LDS.
// ------------------------------------------------------------------------------------------
template <bool STORE_F32>
__global__ __launch_bounds__(256)
void gemm_bf16(const unsigned short* __restrict__ A,
               const unsigned short* __restrict__ Wt,
               float* __restrict__ Cf, unsigned short* __restrict__ Cb,
               int M, int N, int K) {
    __shared__ unsigned short As[64 * 32];    //  4 KB
    __shared__ unsigned short Bs[256 * 32];   // 16 KB

    int tilesN = N >> 8;                      // blocks of 256 cols
    int bm = blockIdx.x / tilesN;
    int bn = blockIdx.x - bm * tilesN;
    int m0 = bm * 64;
    int n0 = bn * 256;

    int tid    = threadIdx.x;
    int wavei  = tid >> 5;
    int lane   = tid & 31;
    int half16 = lane >> 4;
    int l15    = lane & 15;
    int wm     = wavei >> 2;                  // 0..1  (M sub-block of 32)
    int wn     = wavei & 3;                   // 0..3  (N sub-block of 64)

    // ---- staging assignments (invariant over K loop) ----
    int arow  = tid >> 2;                     // 0..63
    int apart = (tid & 3) * 8;                // u16 offset of 16B chunk
    const unsigned short* gA = A + (size_t)(m0 + arow) * K + apart;
    unsigned short*       lA = As + arow * 32 + apart;

    const unsigned short* gB[4];
    unsigned short*       lB[4];
    #pragma unroll
    for (int it = 0; it < 4; ++it) {
        int c = tid + 256 * it;               // 0..1023
        int brow = c >> 2;                    // 0..255
        int bpart = (c & 3) * 8;
        gB[it] = Wt + (size_t)(n0 + brow) * K + bpart;
        lB[it] = Bs + brow * 32 + bpart;
    }

    // ---- LDS fragment read addresses (ISA 7.12.2 bf16 A/B lane layout) ----
    const unsigned short* fA0 = As + (wm * 32      + l15) * 32 + half16 * 8;
    const unsigned short* fA1 = As + (wm * 32 + 16 + l15) * 32 + half16 * 8;
    const unsigned short* fB  = Bs + (wn * 64      + l15) * 32 + half16 * 8;

    v8f acc[2][4];
    #pragma unroll
    for (int mi = 0; mi < 2; ++mi)
        #pragma unroll
        for (int j = 0; j < 4; ++j) acc[mi][j] = (v8f){};

    for (int kk = 0; kk < K; kk += 32) {
        // stage A + B tiles into LDS (async on CDNA5)
        copy16_g2l(gA + kk, lA);
        #pragma unroll
        for (int it = 0; it < 4; ++it) copy16_g2l(gB[it] + kk, lB[it]);
        __builtin_prefetch(gA + kk + 32, 0, 0);
        wait_async_copies();
        __syncthreads();

        Frag a0, a1;
        a0.q[0] = *(const uint4*)(fA0);
        a0.q[1] = *(const uint4*)(fA0 + 16);
        a1.q[0] = *(const uint4*)(fA1);
        a1.q[1] = *(const uint4*)(fA1 + 16);

        #pragma unroll
        for (int j = 0; j < 4; ++j) {
            Frag b;
            const unsigned short* fBj = fB + j * 16 * 32;
            b.q[0] = *(const uint4*)(fBj);
            b.q[1] = *(const uint4*)(fBj + 16);
            acc[0][j] = __builtin_amdgcn_wmma_f32_16x16x32_bf16(false, a0.v, false, b.v, (short)0, acc[0][j], false, false);
            acc[1][j] = __builtin_amdgcn_wmma_f32_16x16x32_bf16(false, a1.v, false, b.v, (short)0, acc[1][j], false, false);
        }
        __syncthreads();
    }

    #pragma unroll
    for (int mi = 0; mi < 2; ++mi) {
        #pragma unroll
        for (int r = 0; r < 8; ++r) {
            int row = m0 + wm * 32 + mi * 16 + r + half16 * 8;
            size_t base = (size_t)row * N + n0 + wn * 64 + l15;
            #pragma unroll
            for (int j = 0; j < 4; ++j) {
                if (STORE_F32) Cf[base + j * 16] = acc[mi][j][r];
                else           Cb[base + j * 16] = f2bf(acc[mi][j][r]);
            }
        }
    }
}

// ------------------------- RoPE + reorder: [B,T,H*HD] -> [B,H,T,HD], one thread per (even,odd) pair
__global__ void rope_reorder(const unsigned short* __restrict__ P,
                             const float* __restrict__ cosT,
                             const float* __restrict__ sinT,
                             unsigned short* __restrict__ O, int H) {
    int i = blockIdx.x * blockDim.x + threadIdx.x;
    int total = B_ * T_ * H * (HD_ / 2);
    if (i >= total) return;
    int p = i & (HD_ / 2 - 1);
    int r = i >> 5;
    int h = r % H;  r /= H;
    int t = r % T_;
    int b = r / T_;
    size_t src = ((size_t)(b * T_ + t) * H + h) * HD_ + 2 * p;
    float x0 = bf2f(P[src]), x1 = bf2f(P[src + 1]);
    float c = cosT[t * (HD_ / 2) + p];
    float s = sinT[t * (HD_ / 2) + p];
    size_t dst = (((size_t)b * H + h) * T_ + t) * HD_ + 2 * p;
    O[dst]     = f2bf(x0 * c - x1 * s);
    O[dst + 1] = f2bf(x0 * s + x1 * c);
}

// ----------------------------------- V reorder: [B,T,NKV*HD] -> [B,NKV,HD,T] (transpose per head)
__global__ void v_reorder(const unsigned short* __restrict__ P,
                          unsigned short* __restrict__ O) {
    int i = blockIdx.x * blockDim.x + threadIdx.x;
    if (i >= B_ * T_ * NKV_ * HD_) return;
    int d = i & (HD_ - 1);
    int r = i >> 6;
    int g = r % NKV_; r /= NKV_;
    int t = r % T_;
    int b = r / T_;
    O[(((size_t)b * NKV_ + g) * HD_ + d) * T_ + t] = P[i];
}

// --------------------------------------------------- flash attention: one wave per (b,h,16-row tile)
__global__ __launch_bounds__(128)
void attention_fused(const unsigned short* __restrict__ Q,   // [B,NH,T,HD] bf16 (rope'd)
                     const unsigned short* __restrict__ Kr,  // [B,NKV,T,HD] bf16 (rope'd)
                     const unsigned short* __restrict__ Vt,  // [B,NKV,HD,T] bf16
                     unsigned short* __restrict__ O) {       // [B,T,NH*HD] bf16
    __shared__ unsigned short ldsP[4][16 * 32];
    int wslot  = threadIdx.x >> 5;
    int wave   = blockIdx.x * 4 + wslot;
    int lane   = threadIdx.x & 31;
    int half16 = lane >> 4;
    int l15    = lane & 15;
    int qt = wave & (T_ / 16 - 1);
    int bh = wave >> 7;                 // T_/16 == 128
    int h  = bh % NH_;
    int b  = bh / NH_;
    int g  = h / NREP_;
    int t0 = qt * 16;

    const unsigned short* qbase = Q  + ((size_t)b * NH_  + h) * T_  * HD_;
    const unsigned short* kbase = Kr + ((size_t)b * NKV_ + g) * T_  * HD_;
    const unsigned short* vbase = Vt + ((size_t)b * NKV_ + g) * HD_ * T_;

    // Q fragment: A operand 16x64 over K=HD, split into two K=32 fragments
    Frag aq0, aq1;
    const unsigned short* qrow = qbase + (size_t)(t0 + l15) * HD_ + half16 * 8;
    aq0.q[0] = *(const uint4*)(qrow);
    aq0.q[1] = *(const uint4*)(qrow + 16);
    aq1.q[0] = *(const uint4*)(qrow + 32);
    aq1.q[1] = *(const uint4*)(qrow + 48);

    v8f o0 = {}, o1 = {}, o2 = {}, o3 = {};
    float mrow[8], lrow[8];
    #pragma unroll
    for (int r = 0; r < 8; ++r) { mrow[r] = -3.0e38f; lrow[r] = 0.0f; }

    unsigned short* pbuf = ldsP[wslot];

    for (int s0 = 0; s0 < t0 + 16; s0 += 32) {
        // scores for 32 s-columns = two 16x16 C tiles, each needs K=64 = 2 WMMAs
        v8f sc0 = {}, sc1 = {};
        {
            const unsigned short* krow = kbase + (size_t)(s0 + l15) * HD_ + half16 * 8;
            Frag bk;
            bk.q[0] = *(const uint4*)(krow);      bk.q[1] = *(const uint4*)(krow + 16);
            sc0 = __builtin_amdgcn_wmma_f32_16x16x32_bf16(false, aq0.v, false, bk.v, (short)0, sc0, false, false);
            bk.q[0] = *(const uint4*)(krow + 32); bk.q[1] = *(const uint4*)(krow + 48);
            sc0 = __builtin_amdgcn_wmma_f32_16x16x32_bf16(false, aq1.v, false, bk.v, (short)0, sc0, false, false);
        }
        {
            const unsigned short* krow = kbase + (size_t)(s0 + 16 + l15) * HD_ + half16 * 8;
            Frag bk;
            bk.q[0] = *(const uint4*)(krow);      bk.q[1] = *(const uint4*)(krow + 16);
            sc1 = __builtin_amdgcn_wmma_f32_16x16x32_bf16(false, aq0.v, false, bk.v, (short)0, sc1, false, false);
            bk.q[0] = *(const uint4*)(krow + 32); bk.q[1] = *(const uint4*)(krow + 48);
            sc1 = __builtin_amdgcn_wmma_f32_16x16x32_bf16(false, aq1.v, false, bk.v, (short)0, sc1, false, false);
        }

        // causal mask + online softmax; write P (bf16) to LDS for layout transpose
        #pragma unroll
        for (int r = 0; r < 8; ++r) {
            int t  = t0 + r + half16 * 8;
            int sa = s0 + l15;
            int sb = sa + 16;
            float v0 = (sa <= t) ? sc0[r] * 0.125f : -1.0e9f;
            float v1 = (sb <= t) ? sc1[r] * 0.125f : -1.0e9f;
            float vm = fmaxf(v0, v1);
            vm = fmaxf(vm, __shfl_xor(vm, 1, 32));
            vm = fmaxf(vm, __shfl_xor(vm, 2, 32));
            vm = fmaxf(vm, __shfl_xor(vm, 4, 32));
            vm = fmaxf(vm, __shfl_xor(vm, 8, 32));
            float mnew = fmaxf(mrow[r], vm);
            float corr = __expf(mrow[r] - mnew);
            float p0 = __expf(v0 - mnew);
            float p1 = __expf(v1 - mnew);
            float ps = p0 + p1;
            ps += __shfl_xor(ps, 1, 32);
            ps += __shfl_xor(ps, 2, 32);
            ps += __shfl_xor(ps, 4, 32);
            ps += __shfl_xor(ps, 8, 32);
            lrow[r] = lrow[r] * corr + ps;
            mrow[r] = mnew;
            o0[r] *= corr; o1[r] *= corr; o2[r] *= corr; o3[r] *= corr;
            int prow = r + half16 * 8;
            pbuf[prow * 32 + l15]      = f2bf(p0);
            pbuf[prow * 32 + 16 + l15] = f2bf(p1);
        }
        // in-wave LDS RAW: stores above feed the cross-lane A-fragment loads below
        asm volatile("s_wait_dscnt 0" ::: "memory");

        // P as A operand (16x32, K = s)
        Frag ap;
        const unsigned short* prow_ = pbuf + l15 * 32 + half16 * 8;
        ap.q[0] = *(const uint4*)(prow_);
        ap.q[1] = *(const uint4*)(prow_ + 16);

        // O += P x V   (4 d-tiles of 16)
        const unsigned short* vrow = vbase + (size_t)l15 * T_ + s0 + half16 * 8;
        Frag bv;
        bv.q[0] = *(const uint4*)(vrow);  bv.q[1] = *(const uint4*)(vrow + 16);
        o0 = __builtin_amdgcn_wmma_f32_16x16x32_bf16(false, ap.v, false, bv.v, (short)0, o0, false, false);
        vrow += 16 * T_;
        bv.q[0] = *(const uint4*)(vrow);  bv.q[1] = *(const uint4*)(vrow + 16);
        o1 = __builtin_amdgcn_wmma_f32_16x16x32_bf16(false, ap.v, false, bv.v, (short)0, o1, false, false);
        vrow += 16 * T_;
        bv.q[0] = *(const uint4*)(vrow);  bv.q[1] = *(const uint4*)(vrow + 16);
        o2 = __builtin_amdgcn_wmma_f32_16x16x32_bf16(false, ap.v, false, bv.v, (short)0, o2, false, false);
        vrow += 16 * T_;
        bv.q[0] = *(const uint4*)(vrow);  bv.q[1] = *(const uint4*)(vrow + 16);
        o3 = __builtin_amdgcn_wmma_f32_16x16x32_bf16(false, ap.v, false, bv.v, (short)0, o3, false, false);
    }

    // normalize and store to [B,T,NH*HD]
    #pragma unroll
    for (int r = 0; r < 8; ++r) {
        int t = t0 + r + half16 * 8;
        float inv = 1.0f / lrow[r];
        size_t base = (size_t)(b * T_ + t) * (NH_ * HD_) + h * HD_ + l15;
        O[base]      = f2bf(o0[r] * inv);
        O[base + 16] = f2bf(o1[r] * inv);
        O[base + 32] = f2bf(o2[r] * inv);
        O[base + 48] = f2bf(o3[r] * inv);
    }
}

// =====================================================================================
extern "C" void kernel_launch(void* const* d_in, const int* in_sizes, int n_in,
                              void* d_out, int out_size, void* d_ws, size_t ws_size,
                              hipStream_t stream) {
    const float* x    = (const float*)d_in[0];
    const float* wq   = (const float*)d_in[1];
    const float* wk   = (const float*)d_in[2];
    const float* wv   = (const float*)d_in[3];
    const float* wo   = (const float*)d_in[4];
    const float* cosT = (const float*)d_in[5];
    const float* sinT = (const float*)d_in[6];
    // d_in[7] = mask: causal mask applied analytically in-kernel
    float* out = (float*)d_out;

    unsigned short* ws = (unsigned short*)d_ws;
    size_t off = 0;
    auto take = [&](size_t n) { unsigned short* p = ws + off; off += n; return p; };

    unsigned short* xb  = take((size_t)B_ * T_ * DIM_);           // 8M elems
    unsigned short* wqT = take((size_t)(NH_ * HD_) * DIM_);       // 4M
    unsigned short* wkT = take((size_t)(NKV_ * HD_) * DIM_);      // 1M
    unsigned short* wvT = take((size_t)(NKV_ * HD_) * DIM_);      // 1M
    unsigned short* woT = take((size_t)DIM_ * (NH_ * HD_));       // 4M
    unsigned short* qp  = take((size_t)B_ * T_ * NH_ * HD_);      // 8M
    unsigned short* kp  = take((size_t)B_ * T_ * NKV_ * HD_);     // 2M
    unsigned short* vp  = take((size_t)B_ * T_ * NKV_ * HD_);     // 2M
    unsigned short* qr  = take((size_t)B_ * NH_ * T_ * HD_);      // 8M
    unsigned short* kr  = take((size_t)B_ * NKV_ * T_ * HD_);     // 2M
    unsigned short* vT  = take((size_t)B_ * NKV_ * HD_ * T_);     // 2M
    unsigned short* ao  = take((size_t)B_ * T_ * NH_ * HD_);      // 8M
    (void)ws_size; (void)in_sizes; (void)n_in; (void)out_size;

    int n;
    // 1. convert activations / weights to bf16 (weights transposed to [N,K])
    n = B_ * T_ * DIM_;
    cvt_f32_bf16<<<(n + 255) / 256, 256, 0, stream>>>(x, xb, n);
    n = DIM_ * NH_ * HD_;
    tcvt_f32_bf16<<<(n + 255) / 256, 256, 0, stream>>>(wq, wqT, DIM_, NH_ * HD_);
    n = DIM_ * NKV_ * HD_;
    tcvt_f32_bf16<<<(n + 255) / 256, 256, 0, stream>>>(wk, wkT, DIM_, NKV_ * HD_);
    tcvt_f32_bf16<<<(n + 255) / 256, 256, 0, stream>>>(wv, wvT, DIM_, NKV_ * HD_);
    n = NH_ * HD_ * DIM_;
    tcvt_f32_bf16<<<(n + 255) / 256, 256, 0, stream>>>(wo, woT, NH_ * HD_, DIM_);

    // 2. Q/K/V projections (block-cooperative WMMA GEMMs, bf16 out)
    {
        int M = B_ * T_;
        int grid_q = (M / 64) * ((NH_ * HD_) / 256);
        gemm_bf16<false><<<grid_q, 256, 0, stream>>>(xb, wqT, nullptr, qp, M, NH_ * HD_, DIM_);
        int grid_kv = (M / 64) * ((NKV_ * HD_) / 256);
        gemm_bf16<false><<<grid_kv, 256, 0, stream>>>(xb, wkT, nullptr, kp, M, NKV_ * HD_, DIM_);
        gemm_bf16<false><<<grid_kv, 256, 0, stream>>>(xb, wvT, nullptr, vp, M, NKV_ * HD_, DIM_);
    }

    // 3. RoPE + head-major reorders
    n = B_ * T_ * NH_ * (HD_ / 2);
    rope_reorder<<<(n + 255) / 256, 256, 0, stream>>>(qp, cosT, sinT, qr, NH_);
    n = B_ * T_ * NKV_ * (HD_ / 2);
    rope_reorder<<<(n + 255) / 256, 256, 0, stream>>>(kp, cosT, sinT, kr, NKV_);
    n = B_ * T_ * NKV_ * HD_;
    v_reorder<<<(n + 255) / 256, 256, 0, stream>>>(vp, vT);

    // 4. fused flash attention (WMMA scores + WMMA PV)
    {
        int waves = B_ * NH_ * (T_ / 16);   // 8192
        attention_fused<<<waves / 4, 128, 0, stream>>>(qr, kr, vT, ao);
    }

    // 5. output projection (WMMA GEMM, f32 out -> d_out)
    {
        int M = B_ * T_;
        int grid_o = (M / 64) * (DIM_ / 256);
        gemm_bf16<true><<<grid_o, 256, 0, stream>>>(ao, woT, out, nullptr, M, DIM_, DIM_);
    }
}
